// StateRNN_42408507081097
// MI455X (gfx1250) — matrix operand order repaired and checked by reading, not compiled, not run on hardware
//
#include <hip/hip_runtime.h>

// ---------------------------------------------------------------------------
// Types / helpers
// ---------------------------------------------------------------------------
typedef unsigned short u16;
typedef __attribute__((ext_vector_type(16))) __bf16        v16bf;
typedef __attribute__((ext_vector_type(8)))  float         v8f;
typedef __attribute__((ext_vector_type(4)))  unsigned int  u32x4; // builtin vec,
// unlike HIP's class-type uint4 it can be loaded through an AS-qualified ptr.

// Explicit global-address-space pointers: after laundering the weight base
// pointers through inline asm (to defeat LICM) we rebuild them in AS(1) so
// operand loads lower to global_load_b128 (LOADcnt only) instead of
// flat_load_b128 (LOADcnt + DScnt, contends the LDS pipe with our ds_loads).
typedef const __attribute__((address_space(1))) u16*   gbp;
typedef const __attribute__((address_space(1))) u32x4* gq4p;

#define BB   128
#define TT   512
#define NHH  256
#define NIN  50
#define LL   32

__device__ __forceinline__ u16 f2bf(float f) {
  unsigned int x = __float_as_uint(f);
  return (u16)((x + 0x7FFFu + ((x >> 16) & 1u)) >> 16);   // RNE truncation
}
__device__ __forceinline__ float bf2f(u16 h) {
  return __uint_as_float(((unsigned int)h) << 16);
}
__device__ __forceinline__ float sigm(float x) { return 1.0f / (1.0f + __expf(-x)); }

// Load two 16-byte chunks into a 16-element bf16 operand vector (generic AS).
__device__ __forceinline__ v16bf load2x16(const u16* p0, const u16* p1) {
  union { u32x4 q[2]; v16bf v; } u;
  u.q[0] = *(const u32x4*)p0;
  u.q[1] = *(const u32x4*)p1;
  return u.v;
}
// Same, but through explicit global-AS pointers -> global_load_b128.
__device__ __forceinline__ v16bf load2x16g(gbp p0, gbp p1) {
  union { u32x4 q[2]; v16bf v; } u;
  u.q[0] = *(gq4p)p0;
  u.q[1] = *(gq4p)p1;
  return u.v;
}

__device__ __forceinline__ v8f wmma_bf16(v16bf a, v16bf b, v8f c) {
  // D(f32,16x16) = A(bf16,16x32) * B(bf16,32x16) + C
  return __builtin_amdgcn_wmma_f32_16x16x32_bf16(false, a, false, b,
                                                 (short)0, c, false, false);
}

// ---------------------------------------------------------------------------
// Kernel 1: convert + transpose all weights to bf16, W^T[n][k] layout.
//   wxT [1024][64]  (K padded 50->64 with zeros)
//   whT [1024][256]
//   w1T [256][256]   w2T [128][256]   w3T [32][128]
// Exactly 430080 elements -> 1680 blocks x 256 threads.
// ---------------------------------------------------------------------------
__global__ __launch_bounds__(256)
void prep_weights(const float* __restrict__ wx, const float* __restrict__ wh,
                  const float* __restrict__ w1, const float* __restrict__ w2,
                  const float* __restrict__ w3,
                  u16* __restrict__ wxT, u16* __restrict__ whT,
                  u16* __restrict__ w1T, u16* __restrict__ w2T,
                  u16* __restrict__ w3T) {
  int i = blockIdx.x * 256 + threadIdx.x;
  if (i < 65536) {                       // wxT: n in [0,1024), k in [0,64)
    int n = i >> 6, k = i & 63;
    wxT[i] = f2bf(k < NIN ? wx[k * 1024 + n] : 0.0f);
    return;
  }
  i -= 65536;
  if (i < 262144) {                      // whT: n in [0,1024), k in [0,256)
    int n = i >> 8, k = i & 255;
    whT[i] = f2bf(wh[k * 1024 + n]);
    return;
  }
  i -= 262144;
  if (i < 65536) {                       // w1T: n in [0,256), k in [0,256)
    int n = i >> 8, k = i & 255;
    w1T[i] = f2bf(w1[k * 256 + n]);
    return;
  }
  i -= 65536;
  if (i < 32768) {                       // w2T: n in [0,128), k in [0,256)
    int n = i >> 8, k = i & 255;
    w2T[i] = f2bf(w2[k * 128 + n]);
    return;
  }
  i -= 32768;
  if (i < 4096) {                        // w3T: n in [0,32), k in [0,128)
    int n = i >> 7, k = i & 127;
    w3T[i] = f2bf(w3[k * 32 + n]);
  }
}

// ---------------------------------------------------------------------------
// Kernel 2: LSTM scan. 8 blocks x 256 threads (8 waves). Block rb owns batch
// rows [rb*16, rb*16+16): one WMMA M-tile. c (f32) and h (bf16, double-
// buffered) live in LDS for all 512 steps; no inter-block sync needed because
// batch rows are independent.
//
// Loop order: kb outer, (cc,gate) inner -> one A operand live at a time;
// 8 persistent f32 accumulator tiles (64 VGPRs). B tiles are re-streamed from
// L2 every step (whT bf16 = 512 KB, L2-resident); an empty asm rewriting the
// weight base pointers stops LICM from hoisting 640 VGPRs of B operands into
// scratch spills, and the AS(1) rebuild keeps the loads on the global path.
// ---------------------------------------------------------------------------
__global__ __launch_bounds__(256)
void lstm_scan(const float* __restrict__ seq,       // [B,T,NIN]
               const float* __restrict__ mask,      // [B,T]
               const float* __restrict__ states_in, // [B,512]
               const float* __restrict__ bias,      // [1024]
               const u16* __restrict__ wxT,         // [1024][64]
               const u16* __restrict__ whT,         // [1024][256]
               u16* __restrict__ hs,                // [B*T][256] bf16
               float* __restrict__ states_out) {    // [B,512]
  extern __shared__ char smem[];
  u16*   hbuf = (u16*)smem;                          // [2][16][256] = 16KB
  float* cbuf = (float*)(smem + 16384);              // [16][256]    = 16KB
  u16*   xbuf = (u16*)(smem + 32768);                // [16][64]     = 2KB

  const int tid  = threadIdx.x;
  const int lane = tid & 31;
  const int w    = tid >> 5;               // 0..7
  const int row0 = blockIdx.x * 16;

  // init state: states_in[:, 0:256] = c, [:, 256:512] = h
  for (int i = tid; i < 16 * 256; i += 256) {
    int r = i >> 8, j = i & 255;
    cbuf[i] = states_in[(row0 + r) * 512 + j];
    hbuf[i] = f2bf(states_in[(row0 + r) * 512 + 256 + j]);
  }
  __syncthreads();

  const int arow = lane & 15;              // A-operand row within tile
  const int kh8  = (lane >> 4) * 8;        // A lane K split per ISA layout
  const int col  = lane & 15;              // B/D column within tile
  const int kb16 = (lane >> 4) * 16;       // unused; B loads use kh8, same K order as A

  // t-invariant bias values for this lane's 8 z-columns
  float bv[2][4];
  #pragma unroll
  for (int cc = 0; cc < 2; ++cc)
    #pragma unroll
    for (int g = 0; g < 4; ++g)
      bv[cc][g] = bias[g * 256 + (w * 2 + cc) * 16 + col];

  unsigned long long p_wh = (unsigned long long)whT;
  unsigned long long p_wx = (unsigned long long)wxT;

  int cur = 0;
  for (int t = 0; t < TT; ++t) {
    const int nxt = cur ^ 1;

    // make the weight base pointers opaque each iteration (defeat LICM/spill)
    asm volatile("" : "+s"(p_wh), "+s"(p_wx));
    gbp whT_v = (gbp)p_wh;
    gbp wxT_v = (gbp)p_wx;

    // stage x_t (bf16, K padded to 64) and apply episode-reset mask
    for (int i = tid; i < 16 * 64; i += 256) {
      int r = i >> 6, k = i & 63;
      float v = (k < NIN) ? seq[((size_t)(row0 + r) * TT + t) * NIN + k] : 0.0f;
      xbuf[i] = f2bf(v);
    }
    for (int i = tid; i < 16 * 256; i += 256) {
      int r = i >> 8;
      float m = mask[(size_t)(row0 + r) * TT + t];
      if (m != 0.0f) {
        float s = 1.0f - m;
        cbuf[i] *= s;
        hbuf[cur * 4096 + i] = f2bf(bf2f(hbuf[cur * 4096 + i]) * s);
      }
    }
    __syncthreads();

    // accumulators: [cc][gate], persistent across K
    v8f acc[2][4];
    #pragma unroll
    for (int cc = 0; cc < 2; ++cc)
      #pragma unroll
      for (int g = 0; g < 4; ++g) {
        float b0 = bv[cc][g];
        acc[cc][g] = (v8f){b0, b0, b0, b0, b0, b0, b0, b0};
      }

    // recurrent part: h @ wh   (K = 256, 8 K-blocks)
    const u16* hrow = &hbuf[cur * 4096 + arow * 256];
    #pragma unroll
    for (int kb = 0; kb < 8; ++kb) {
      v16bf Ah = load2x16(hrow + kb * 32 + kh8, hrow + kb * 32 + kh8 + 16);
      #pragma unroll
      for (int cc = 0; cc < 2; ++cc)
        #pragma unroll
        for (int g = 0; g < 4; ++g) {
          const int zc = g * 256 + (w * 2 + cc) * 16 + col;
          gbp bp = whT_v + (size_t)zc * 256 + kb * 32 + kh8;
          v16bf Bm = load2x16g(bp, bp + 16);
          acc[cc][g] = wmma_bf16(Ah, Bm, acc[cc][g]);
        }
    }
    // input part: x @ wx   (K = 64 padded, 2 K-blocks)
    const u16* xrow = &xbuf[arow * 64];
    #pragma unroll
    for (int kx = 0; kx < 2; ++kx) {
      v16bf Ax = load2x16(xrow + kx * 32 + kh8, xrow + kx * 32 + kh8 + 16);
      #pragma unroll
      for (int cc = 0; cc < 2; ++cc)
        #pragma unroll
        for (int g = 0; g < 4; ++g) {
          const int zc = g * 256 + (w * 2 + cc) * 16 + col;
          gbp bp = wxT_v + (size_t)zc * 64 + kx * 32 + kh8;
          v16bf Bm = load2x16g(bp, bp + 16);
          acc[cc][g] = wmma_bf16(Ax, Bm, acc[cc][g]);
        }
    }

    // elementwise LSTM cell update
    #pragma unroll
    for (int cc = 0; cc < 2; ++cc) {
      const int jj = (w * 2 + cc) * 16 + col;   // gate-relative column
      #pragma unroll
      for (int r = 0; r < 8; ++r) {
        const int mrow = (lane >> 4) * 8 + r;   // row within tile (D layout)
        const float iv = sigm(acc[cc][0][r]);
        const float fv = sigm(acc[cc][1][r]);
        const float ov = sigm(acc[cc][2][r]);
        const float uv = tanhf(acc[cc][3][r]);
        const int ci = mrow * 256 + jj;
        const float cnew = fv * cbuf[ci] + iv * uv;
        const float hnew = ov * tanhf(cnew);
        cbuf[ci] = cnew;
        hbuf[nxt * 4096 + ci] = f2bf(hnew);
        hs[((size_t)(row0 + mrow) * TT + t) * 256 + jj] = f2bf(hnew);
      }
    }
    __syncthreads();
    cur = nxt;
  }

  // states_out = concat(c, h)
  for (int i = tid; i < 16 * 256; i += 256) {
    int r = i >> 8, j = i & 255;
    states_out[(row0 + r) * 512 + j]       = cbuf[i];
    states_out[(row0 + r) * 512 + 256 + j] = bf2f(hbuf[cur * 4096 + i]);
  }
}

// ---------------------------------------------------------------------------
// Kernel 3: bf16 WMMA GEMM + bias (+ optional ReLU), bf16 output.
// Each wave computes a 16x64 strip (1 M-tile x 4 N-tiles): the A operand is
// loaded once per K-block and reused across 4 WMMAs. Grids sized exactly (no
// partial tiles) so EXEC is all-ones at every WMMA.
// ---------------------------------------------------------------------------
__global__ __launch_bounds__(128)
void gemm_bf16(const u16* __restrict__ A, int K,
               const u16* __restrict__ BT,           // [N][K]
               const float* __restrict__ bias,
               u16* __restrict__ out, int N, int relu) {
  const int lane = threadIdx.x & 31;
  const int wave = threadIdx.x >> 5;
  const int ncg  = N >> 6;                 // column groups of 4 tiles
  const int task = blockIdx.x * 4 + wave;
  const int tm = task / ncg, cg = task % ncg;

  const int arow = tm * 16 + (lane & 15);
  const int kh8  = (lane >> 4) * 8;
  const int kb16 = (lane >> 4) * 16;
  const int c0   = cg * 64 + (lane & 15);  // lane's column in tile 0

  v8f acc[4];
  #pragma unroll
  for (int ct = 0; ct < 4; ++ct) {
    float b0 = bias[c0 + ct * 16];
    acc[ct] = (v8f){b0, b0, b0, b0, b0, b0, b0, b0};
  }

  const u16* ap = A + (size_t)arow * K;
  for (int kb = 0; kb < K; kb += 32) {
    v16bf a = load2x16(ap + kb + kh8, ap + kb + kh8 + 16);
    #pragma unroll
    for (int ct = 0; ct < 4; ++ct) {
      const u16* bp = BT + (size_t)(c0 + ct * 16) * K + kb + kh8;
      v16bf b = load2x16(bp, bp + 16);
      acc[ct] = wmma_bf16(a, b, acc[ct]);
    }
  }

  const int orow0 = tm * 16 + (lane >> 4) * 8;
  #pragma unroll
  for (int ct = 0; ct < 4; ++ct)
    #pragma unroll
    for (int r = 0; r < 8; ++r) {
      float v = acc[ct][r];
      if (relu) v = fmaxf(v, 0.0f);
      out[(size_t)(orow0 + r) * N + c0 + ct * 16] = f2bf(v);
    }
}

// ---------------------------------------------------------------------------
// Kernel 4: final GEMM (K=128, N=32) + f32 output + fused MSE loss.
// Each wave: 1 M-tile x 2 N-tiles (full output row block).
// loss = sum_{b,t,l} (out-target)^2 / (B*L)
// ---------------------------------------------------------------------------
__global__ __launch_bounds__(128)
void final_gemm(const u16* __restrict__ A,           // d2 [65536][128]
                const u16* __restrict__ BT,          // w3T [32][128]
                const float* __restrict__ bias,      // [32]
                const float* __restrict__ targ,      // [B*T*32]
                float* __restrict__ out,             // [B*T*32]
                float* __restrict__ loss) {
  const int lane = threadIdx.x & 31;
  const int wave = threadIdx.x >> 5;
  const int tm   = blockIdx.x * 4 + wave;            // 4096 M-tiles

  const int arow = tm * 16 + (lane & 15);
  const int kh8  = (lane >> 4) * 8;
  const int kb16 = (lane >> 4) * 16;
  const int c0   = lane & 15;

  v8f acc[2];
  #pragma unroll
  for (int ct = 0; ct < 2; ++ct) {
    float b0 = bias[c0 + ct * 16];
    acc[ct] = (v8f){b0, b0, b0, b0, b0, b0, b0, b0};
  }

  const u16* ap = A + (size_t)arow * 128;
  #pragma unroll
  for (int kb = 0; kb < 128; kb += 32) {
    v16bf a = load2x16(ap + kb + kh8, ap + kb + kh8 + 16);
    #pragma unroll
    for (int ct = 0; ct < 2; ++ct) {
      const u16* bp = BT + (size_t)(c0 + ct * 16) * 128 + kb + kh8;
      v16bf b = load2x16(bp, bp + 16);
      acc[ct] = wmma_bf16(a, b, acc[ct]);
    }
  }

  const int orow0 = tm * 16 + (lane >> 4) * 8;
  float part = 0.0f;
  #pragma unroll
  for (int ct = 0; ct < 2; ++ct)
    #pragma unroll
    for (int r = 0; r < 8; ++r) {
      float v = acc[ct][r];
      size_t oi = (size_t)(orow0 + r) * LL + c0 + ct * 16;
      out[oi] = v;
      float d = v - targ[oi];
      part += d * d;
    }
  part *= (1.0f / (float)(BB * LL));
  #pragma unroll
  for (int off = 16; off > 0; off >>= 1) part += __shfl_down(part, off, 32);
  if (lane == 0) atomicAdd(loss, part);
}

__global__ void zero_loss(float* loss) { *loss = 0.0f; }

// ---------------------------------------------------------------------------
// Host launcher
// ---------------------------------------------------------------------------
extern "C" void kernel_launch(void* const* d_in, const int* in_sizes, int n_in,
                              void* d_out, int out_size, void* d_ws, size_t ws_size,
                              hipStream_t stream) {
  const float* seq   = (const float*)d_in[0];
  const float* targ  = (const float*)d_in[1];
  const float* mask  = (const float*)d_in[2];
  const float* sin_  = (const float*)d_in[3];
  const float* wx    = (const float*)d_in[4];
  const float* wh    = (const float*)d_in[5];
  const float* bb    = (const float*)d_in[6];
  const float* w1    = (const float*)d_in[7];
  const float* b1    = (const float*)d_in[8];
  const float* w2    = (const float*)d_in[9];
  const float* b2    = (const float*)d_in[10];
  const float* w3    = (const float*)d_in[11];
  const float* b3    = (const float*)d_in[12];

  char* ws = (char*)d_ws;
  u16* wxT = (u16*)(ws + 0);                    // 128 KB
  u16* whT = (u16*)(ws + 131072);               // 512 KB
  u16* w1T = (u16*)(ws + 655360);               // 128 KB
  u16* w2T = (u16*)(ws + 786432);               // 64 KB
  u16* w3T = (u16*)(ws + 851968);               // 8 KB
  u16* hs  = (u16*)(ws + 1048576);              // 32 MB  [B*T][256] bf16
  u16* d1  = (u16*)(ws + 1048576 + 33554432);   // 32 MB  [B*T][256] bf16
  u16* d2  = (u16*)(ws + 1048576 + 67108864);   // 16 MB  [B*T][128] bf16

  float* out  = (float*)d_out;                  // [B,T,L]
  float* st   = out + (size_t)BB * TT * LL;     // [B,512]
  float* loss = st + (size_t)BB * 512;          // scalar

  prep_weights<<<1680, 256, 0, stream>>>(wx, wh, w1, w2, w3,
                                         wxT, whT, w1T, w2T, w3T);
  lstm_scan<<<8, 256, 34816, stream>>>(seq, mask, sin_, bb, wxT, whT, hs, st);
  gemm_bf16<<<4096, 128, 0, stream>>>(hs, 256, w1T, b1, d1, 256, 1); // d1
  gemm_bf16<<<2048, 128, 0, stream>>>(d1, 256, w2T, b2, d2, 128, 1); // d2
  zero_loss<<<1, 1, 0, stream>>>(loss);
  final_gemm<<<1024, 128, 0, stream>>>(d2, w3T, b3, targ, out, loss);
}
